// DirectedMPNN_21131239096638
// MI455X (gfx1250) — matrix operand order repaired and checked by reading, not compiled, not run on hardware
//
#include <hip/hip_runtime.h>
#include <hip/hip_bf16.h>

typedef __attribute__((ext_vector_type(16))) _Float16 v16h;
typedef __attribute__((ext_vector_type(8)))  float    v8f;
typedef __attribute__((ext_vector_type(4)))  float    f32x4;

#define D_H 128

__device__ __forceinline__ float lrelu(float v){ return v >= 0.f ? v : 0.01f * v; }

// split f32 into hi/lo f16 pair: x ~= (float)hi + (float)lo
__device__ __forceinline__ void split_f16(float x, _Float16& hi, _Float16& lo){
  hi = (_Float16)x;
  lo = (_Float16)(x - (float)hi);
}

// ---------------------------------------------------------------- utilities
__global__ void k_zero_f32(float* __restrict__ p, int n4){   // n4 = n/4
  int i = blockIdx.x * blockDim.x + threadIdx.x;
  if (i < n4){
    f32x4 z = {0.f, 0.f, 0.f, 0.f};
    ((f32x4*)p)[i] = z;
  }
}

// segment-sum: agg[idx[e], :] += h[e, :]   (one thread per (e, 4-col) chunk)
__global__ void k_scatter_add(const float* __restrict__ h, const int* __restrict__ idx,
                              float* __restrict__ agg, int E){
  int t = blockIdx.x * blockDim.x + threadIdx.x;
  int total = E * (D_H / 4);
  if (t < total){
    int e = t >> 5;
    int d = (t & 31) * 4;
    f32x4 v = *(const f32x4*)(h + (size_t)e * D_H + d);
    float* dst = agg + (size_t)idx[e] * D_H + d;
    atomicAdd(dst + 0, v[0]);
    atomicAdd(dst + 1, v[1]);
    atomicAdd(dst + 2, v[2]);
    atomicAdd(dst + 3, v[3]);
  }
}

// ---------------------------------------------------------------- init GEMM
// h0 = lrelu(concat_feats[E,147] @ W_init^T + b_init)
// 8 waves/block; each wave owns a 16x128 stripe. 3-term f16-split WMMA:
//   x@W ~= xh@Wh + xl@Wh + xh@Wl   (f32-grade accuracy, f16 WMMA rate)
__global__ __launch_bounds__(256)
void k_gemm_init(const float* __restrict__ cf,     // [E,147]
                 const float* __restrict__ W,      // [128,147] row-major
                 const float* __restrict__ bias,   // [128]
                 float* __restrict__ hout,         // [E,128]
                 int E)
{
  const int KSRC = 147, KPAD = 160;
  __shared__ __align__(32) _Float16 ldsWh[D_H * KPAD];
  __shared__ __align__(32) _Float16 ldsWl[D_H * KPAD];
  for (int i = threadIdx.x; i < D_H * KPAD; i += 256){
    int r = i / KPAD, k = i - r * KPAD;
    float w = (k < KSRC) ? W[r * KSRC + k] : 0.f;
    _Float16 hi, lo; split_f16(w, hi, lo);
    ldsWh[i] = hi; ldsWl[i] = lo;
  }
  __syncthreads();

  const int lane = threadIdx.x & 31;
  const int wv   = threadIdx.x >> 5;
  const int half = lane >> 4;
  const int l16  = lane & 15;
  const int row0 = blockIdx.x * 128 + wv * 16;

  const int  eA    = row0 + l16;       // A row owned by this lane (both halves)
  const bool rowok = eA < E;
  const float* pa  = cf + (size_t)eA * KSRC;

  v8f acc[8] = {};
  for (int k0 = 0; k0 < KPAD; k0 += 32){
    v16h ah = {}, al = {};
    if (rowok){
      int ks0 = k0 + half * 16;
      int ks1 = k0 + half * 16 + 8;
      #pragma unroll
      for (int j = 0; j < 8; ++j){
        int ka = ks0 + j, kb = ks1 + j;
        float xa = (ka < KSRC) ? pa[ka] : 0.f;
        float xb = (kb < KSRC) ? pa[kb] : 0.f;
        _Float16 h0, l0, h1, l1;
        split_f16(xa, h0, l0); split_f16(xb, h1, l1);
        ah[j] = h0; al[j] = l0; ah[8 + j] = h1; al[8 + j] = l1;
      }
    }
    #pragma unroll
    for (int nt = 0; nt < 8; ++nt){
      int boff = (nt * 16 + l16) * KPAD + k0 + half * 16;
      v16h bh = *(const v16h*)&ldsWh[boff];
      v16h bl = *(const v16h*)&ldsWl[boff];
      acc[nt] = __builtin_amdgcn_wmma_f32_16x16x32_f16(false, ah, false, bh, (short)0, acc[nt], false, false);
      acc[nt] = __builtin_amdgcn_wmma_f32_16x16x32_f16(false, al, false, bh, (short)0, acc[nt], false, false);
      acc[nt] = __builtin_amdgcn_wmma_f32_16x16x32_f16(false, ah, false, bl, (short)0, acc[nt], false, false);
    }
  }

  #pragma unroll
  for (int nt = 0; nt < 8; ++nt){
    int col = nt * 16 + l16;
    float bv = bias[col];
    #pragma unroll
    for (int r = 0; r < 8; ++r){
      int e = row0 + r + half * 8;
      if (e < E) hout[(size_t)e * D_H + col] = lrelu(acc[nt][r] + bv);
    }
  }
}

// ---------------------------------------------------------------- K=128 GEMM
// MP:    A row e = agg[srt[e],:] - Hp[end[e],:]  (fused message-passing gather)
// SKIP:  out += Hp[e,:]
// out = lrelu(A @ W^T + bias [+ skip]) ; 3-term f16-split WMMA.
template<bool MP, bool SKIP>
__global__ __launch_bounds__(256)
void k_gemm128(const float* __restrict__ A,      // MP: agg [N,128]; else [rows,128]
               const float* __restrict__ Hp,     // h_prev [E,128] (gather + skip src)
               const int* __restrict__ srt,
               const int* __restrict__ endi,
               const float* __restrict__ W,      // [128,128] row-major
               const float* __restrict__ bias,   // [128]
               float* __restrict__ out,          // [rows,128]
               int rows)
{
  __shared__ __align__(32) _Float16 ldsWh[D_H * D_H];
  __shared__ __align__(32) _Float16 ldsWl[D_H * D_H];
  for (int i = threadIdx.x; i < D_H * D_H; i += 256){
    _Float16 hi, lo; split_f16(W[i], hi, lo);
    ldsWh[i] = hi; ldsWl[i] = lo;
  }
  __syncthreads();

  const int lane = threadIdx.x & 31;
  const int wv   = threadIdx.x >> 5;
  const int half = lane >> 4;
  const int l16  = lane & 15;
  const int row0 = blockIdx.x * 128 + wv * 16;
  const int eA   = row0 + l16;
  const bool rowok = eA < rows;

  const float* pa;
  const float* ph = nullptr;
  if (MP){
    int s = rowok ? srt[eA]  : 0;
    int t = rowok ? endi[eA] : 0;
    pa = A  + (size_t)s * D_H;
    ph = Hp + (size_t)t * D_H;
  } else {
    pa = A + (size_t)(rowok ? eA : 0) * D_H;
  }

  // prefetch the skip-connection row re-read in the epilogue (global_prefetch_b8)
  if (SKIP && rowok)
    __builtin_prefetch(Hp + (size_t)eA * D_H + half * 64, 0, 0);

  v8f acc[8] = {};
  #pragma unroll
  for (int k0 = 0; k0 < D_H; k0 += 32){
    v16h ah = {}, al = {};
    if (rowok){
      int ks0 = k0 + half * 16;
      int ks1 = k0 + half * 16 + 8;
      f32x4 u0 = *(const f32x4*)(pa + ks0);
      f32x4 u1 = *(const f32x4*)(pa + ks0 + 4);
      f32x4 u2 = *(const f32x4*)(pa + ks1);
      f32x4 u3 = *(const f32x4*)(pa + ks1 + 4);
      if (MP){
        u0 -= *(const f32x4*)(ph + ks0);
        u1 -= *(const f32x4*)(ph + ks0 + 4);
        u2 -= *(const f32x4*)(ph + ks1);
        u3 -= *(const f32x4*)(ph + ks1 + 4);
      }
      #pragma unroll
      for (int j = 0; j < 4; ++j){
        _Float16 hi, lo;
        split_f16(u0[j], hi, lo); ah[j]      = hi; al[j]      = lo;
        split_f16(u1[j], hi, lo); ah[4 + j]  = hi; al[4 + j]  = lo;
        split_f16(u2[j], hi, lo); ah[8 + j]  = hi; al[8 + j]  = lo;
        split_f16(u3[j], hi, lo); ah[12 + j] = hi; al[12 + j] = lo;
      }
    }
    #pragma unroll
    for (int nt = 0; nt < 8; ++nt){
      int boff = (nt * 16 + l16) * D_H + k0 + half * 16;
      v16h bh = *(const v16h*)&ldsWh[boff];
      v16h bl = *(const v16h*)&ldsWl[boff];
      acc[nt] = __builtin_amdgcn_wmma_f32_16x16x32_f16(false, ah, false, bh, (short)0, acc[nt], false, false);
      acc[nt] = __builtin_amdgcn_wmma_f32_16x16x32_f16(false, al, false, bh, (short)0, acc[nt], false, false);
      acc[nt] = __builtin_amdgcn_wmma_f32_16x16x32_f16(false, ah, false, bl, (short)0, acc[nt], false, false);
    }
  }

  #pragma unroll
  for (int nt = 0; nt < 8; ++nt){
    int col = nt * 16 + l16;
    float bv = bias[col];
    #pragma unroll
    for (int r = 0; r < 8; ++r){
      int e = row0 + r + half * 8;
      if (e < rows){
        float v = acc[nt][r] + bv;
        if (SKIP) v += Hp[(size_t)e * D_H + col];
        out[(size_t)e * D_H + col] = lrelu(v);
      }
    }
  }
}

// ---------------------------------------------------------------- pooling
// L2-normalize each atom row, atomically accumulate into its graph slot.
// One wave32 per node; each lane owns 4 columns.
__global__ void k_graph_pool(const float* __restrict__ atom, const int* __restrict__ batch,
                             float* __restrict__ hg, int N){
  int node = blockIdx.x * 8 + (threadIdx.x >> 5);
  int lane = threadIdx.x & 31;
  if (node >= N) return;
  const float* row = atom + (size_t)node * D_H;
  float v[4]; float s = 0.f;
  #pragma unroll
  for (int j = 0; j < 4; ++j){ v[j] = row[lane + 32 * j]; s += v[j] * v[j]; }
  #pragma unroll
  for (int off = 16; off > 0; off >>= 1) s += __shfl_xor(s, off, 32);
  float r = 1.f / fmaxf(sqrtf(s), 1e-12f);
  int g = batch[node];
  #pragma unroll
  for (int j = 0; j < 4; ++j)
    atomicAdd(&hg[(size_t)g * D_H + lane + 32 * j], v[j] * r);
}

// out[g] = dot(hg[g,:], W_out) + b_out ; one wave32 per graph
__global__ void k_out(const float* __restrict__ hg, const float* __restrict__ Wout,
                      const float* __restrict__ bout, float* __restrict__ out, int G){
  int g = blockIdx.x * 8 + (threadIdx.x >> 5);
  int lane = threadIdx.x & 31;
  if (g >= G) return;
  const float* row = hg + (size_t)g * D_H;
  float s = 0.f;
  #pragma unroll
  for (int j = 0; j < 4; ++j) s += row[lane + 32 * j] * Wout[lane + 32 * j];
  #pragma unroll
  for (int off = 16; off > 0; off >>= 1) s += __shfl_xor(s, off, 32);
  if (lane == 0) out[g] = s + bout[0];
}

// ---------------------------------------------------------------- launch
extern "C" void kernel_launch(void* const* d_in, const int* in_sizes, int n_in,
                              void* d_out, int out_size, void* d_ws, size_t ws_size,
                              hipStream_t stream){
  const float* cf     = (const float*)d_in[1];
  const int*   srt    = (const int*)  d_in[2];
  const int*   endi   = (const int*)  d_in[3];
  const int*   batch  = (const int*)  d_in[5];
  const float* W_init = (const float*)d_in[6];
  const float* b_init = (const float*)d_in[7];
  const float* W_h1   = (const float*)d_in[8];
  const float* b_h1   = (const float*)d_in[9];
  const float* W_mol  = (const float*)d_in[10];
  const float* b_mol  = (const float*)d_in[11];
  const float* W_out  = (const float*)d_in[12];
  const float* b_out  = (const float*)d_in[13];

  const int N = in_sizes[0] / 133;    // x is [N,133] (only its shape is used)
  const int E = in_sizes[2];
  const int G = out_size - N * D_H;   // d_out = [out (G)] ++ [atom_embs (N*128)]

  // workspace: two E x 128 ping-pong buffers, node accumulator, graph accumulator
  float* h_a = (float*)d_ws;
  float* h_b = h_a + (size_t)E * D_H;
  float* agg = h_b + (size_t)E * D_H;
  float* hg  = agg + (size_t)N * D_H;

  float* out_g = (float*)d_out;
  float* atom  = out_g + G;

  const int aggN4     = N * (D_H / 4);
  const int gemmGridE = (E + 127) / 128;
  const int gemmGridN = (N + 127) / 128;
  const int scatGrid  = (E * (D_H / 4) + 255) / 256;

  // h0 = lrelu(cf @ W_init^T + b_init)
  k_gemm_init<<<gemmGridE, 256, 0, stream>>>(cf, W_init, b_init, h_a, E);

  // layer 1: h1 = lrelu(h0 + mp(h0) @ W^T + b)
  k_zero_f32<<<(aggN4 + 255) / 256, 256, 0, stream>>>(agg, aggN4);
  k_scatter_add<<<scatGrid, 256, 0, stream>>>(h_a, endi, agg, E);
  k_gemm128<true, true><<<gemmGridE, 256, 0, stream>>>(agg, h_a, srt, endi, W_h1, b_h1, h_b, E);

  // layer 2
  k_zero_f32<<<(aggN4 + 255) / 256, 256, 0, stream>>>(agg, aggN4);
  k_scatter_add<<<scatGrid, 256, 0, stream>>>(h_b, endi, agg, E);
  k_gemm128<true, true><<<gemmGridE, 256, 0, stream>>>(agg, h_b, srt, endi, W_h1, b_h1, h_a, E);

  // layer 3
  k_zero_f32<<<(aggN4 + 255) / 256, 256, 0, stream>>>(agg, aggN4);
  k_scatter_add<<<scatGrid, 256, 0, stream>>>(h_a, endi, agg, E);
  k_gemm128<true, true><<<gemmGridE, 256, 0, stream>>>(agg, h_a, srt, endi, W_h1, b_h1, h_b, E);

  // h_node = segment_sum(h3 over srt)
  k_zero_f32<<<(aggN4 + 255) / 256, 256, 0, stream>>>(agg, aggN4);
  k_scatter_add<<<scatGrid, 256, 0, stream>>>(h_b, srt, agg, E);

  // atom_embs = lrelu(h_node @ W_mol^T + b_mol)  -> written straight into d_out
  k_gemm128<false, false><<<gemmGridN, 256, 0, stream>>>(agg, nullptr, nullptr, nullptr,
                                                         W_mol, b_mol, atom, N);

  // hg = segment_sum(normalize(atom_embs), batch); out = hg @ W_out^T + b_out
  k_zero_f32<<<(G * (D_H / 4) + 255) / 256, 256, 0, stream>>>(hg, G * (D_H / 4));
  k_graph_pool<<<(N + 7) / 8, 256, 0, stream>>>(atom, batch, hg, N);
  k_out<<<(G + 7) / 8, 256, 0, stream>>>(hg, W_out, b_out, out_g, G);
}
